// MultiHeadAttention_31971736551589
// MI455X (gfx1250) — compile-verified, matrix operand order checked by reading, not run on hardware
//
#include <hip/hip_runtime.h>

typedef __attribute__((ext_vector_type(16))) _Float16 v16h;
typedef __attribute__((ext_vector_type(8)))  float    v8f;

#define B_   32
#define S_   127
#define H_   4
#define D_   64
#define HD_  256
#define R_   (B_ * S_)          // 4064
#define SCALE_ 0.125f
#define NEG_ (-100000.0f)

// ---------------- async global -> LDS staging (CDNA5 path, guarded) ----------------
#if defined(__gfx1250__) && __has_builtin(__builtin_amdgcn_global_load_async_to_lds_b128)
#define USE_ASYNC_LDS 1
typedef int vint4 __attribute__((vector_size(16)));
typedef vint4 __attribute__((address_space(1))) as1_vint4;   // global
typedef vint4 __attribute__((address_space(3))) as3_vint4;   // LDS
#else
#define USE_ASYNC_LDS 0
#endif

__device__ __forceinline__ void async_copy16(const float* g, float* l) {
#if USE_ASYNC_LDS
    __builtin_amdgcn_global_load_async_to_lds_b128((as1_vint4*)g, (as3_vint4*)l, 0, 0);
#else
    *(float4*)l = *(const float4*)g;
#endif
}

__device__ __forceinline__ void async_wait_all() {
#if USE_ASYNC_LDS
    asm volatile("s_wait_asynccnt 0x0" ::: "memory");
#endif
}

// ---------------- weight f32 -> f16 conversion ----------------
__global__ void cvt_f32_f16(const float* __restrict__ src, _Float16* __restrict__ dst, int n) {
    int i = blockIdx.x * blockDim.x + threadIdx.x;
    if (i < n) dst[i] = (_Float16)src[i];
}

// ---------------- WMMA GEMM: C[R x 256] = A[R x 256] @ W^T + bias ----------------
// One wave computes a 16(M) x 64(N) tile: A fragment shared by 4 WMMAs per K-step.
__global__ void gemm_wmma(const float* __restrict__ A, const _Float16* __restrict__ Wh,
                          const float* __restrict__ bias, float* __restrict__ C) {
    const int m0   = blockIdx.x * 16;
    const int n0   = blockIdx.y * 64;
    const int lane = threadIdx.x;        // 0..31, wave32 (EXEC all-1s)
    const int mr   = lane & 15;
    const int hi   = lane >> 4;

    v8f acc0 = {}, acc1 = {}, acc2 = {}, acc3 = {};
    const float*    arow = A  + (size_t)(m0 + mr) * HD_;
    const _Float16* b0   = Wh + (size_t)(n0 +  0 + mr) * HD_;
    const _Float16* b1   = Wh + (size_t)(n0 + 16 + mr) * HD_;
    const _Float16* b2   = Wh + (size_t)(n0 + 32 + mr) * HD_;
    const _Float16* b3   = Wh + (size_t)(n0 + 48 + mr) * HD_;

    for (int kk = 0; kk < HD_; kk += 32) {
        v16h af, bf0, bf1, bf2, bf3;
        // 16-bit A-matrix 16x32 layout: elems 0..7 -> K = 8*hi+e, elems 8..15 -> K = 16+8*hi+e
        #pragma unroll
        for (int e = 0; e < 8; ++e) {
            const int lo = kk + 8 * hi + e;
            const int up = kk + 16 + 8 * hi + e;
            af[e]      = (_Float16)arow[lo];
            af[e + 8]  = (_Float16)arow[up];
            bf0[e] = b0[lo]; bf0[e + 8] = b0[up];
            bf1[e] = b1[lo]; bf1[e + 8] = b1[up];
            bf2[e] = b2[lo]; bf2[e + 8] = b2[up];
            bf3[e] = b3[lo]; bf3[e + 8] = b3[up];
        }
        acc0 = __builtin_amdgcn_wmma_f32_16x16x32_f16(false, af, false, bf0, (short)0, acc0, false, false);
        acc1 = __builtin_amdgcn_wmma_f32_16x16x32_f16(false, af, false, bf1, (short)0, acc1, false, false);
        acc2 = __builtin_amdgcn_wmma_f32_16x16x32_f16(false, af, false, bf2, (short)0, acc2, false, false);
        acc3 = __builtin_amdgcn_wmma_f32_16x16x32_f16(false, af, false, bf3, (short)0, acc3, false, false);
    }

    const float bv0 = bias[n0 +  0 + mr];
    const float bv1 = bias[n0 + 16 + mr];
    const float bv2 = bias[n0 + 32 + mr];
    const float bv3 = bias[n0 + 48 + mr];
    // C/D layout: VGPR j holds M = j + 8*hi, N = lane&15
    #pragma unroll
    for (int j = 0; j < 8; ++j) {
        const int row = m0 + j + 8 * hi;
        float* crow = C + (size_t)row * HD_ + mr;
        crow[n0 +  0] = acc0[j] + bv0;
        crow[n0 + 16] = acc1[j] + bv1;
        crow[n0 + 32] = acc2[j] + bv2;
        crow[n0 + 48] = acc3[j] + bv3;
    }
}

// ---------------- edge-attention core ----------------
__device__ __forceinline__ float dot4(float4 a, float4 b) {
    return a.x * b.x + a.y * b.y + a.z * b.z + a.w * b.w;
}

__global__ void attn_kernel(const float* __restrict__ Qp, const float* __restrict__ Kp,
                            const float* __restrict__ Vp, const int* __restrict__ graph,
                            const float* __restrict__ edge_key,
                            const float* __restrict__ edge_value,
                            const float* __restrict__ edge_query,
                            float* __restrict__ Xa) {
    __shared__ __align__(16) float qrow[HD_];
    __shared__ __align__(16) float ekbuf[S_ * D_];   // edge_key[b,q,:,:]   32 KB
    __shared__ __align__(16) float eqbuf[S_ * D_];   // edge_query[b,:,q,:] 32 KB
    __shared__ float sc[H_][128];
    __shared__ float pr[H_][128];

    const int bq = blockIdx.x;        // 0..R_-1
    const int b  = bq / S_;
    const int q  = bq % S_;
    const int t  = threadIdx.x;       // 0..127

    // stage the projected query row into LDS
    qrow[t]       = Qp[(size_t)bq * HD_ + t];
    qrow[t + 128] = Qp[(size_t)bq * HD_ + t + 128];

    // ---- stage edge_key slice (contiguous 32 KB stream) into LDS ----
    {
        const float* eksrc = edge_key + (((size_t)b * S_ + q) * S_) * D_;
        for (int idx = t; idx < (S_ * D_) / 4; idx += 128)
            async_copy16(eksrc + idx * 4, ekbuf + idx * 4);
    }
    // ---- stage edge_query rows (k-major, 256 B each): 16 lanes per row ----
    {
        const int sub = t & 15;            // lane within row
        for (int k0 = 0; k0 < S_; k0 += 8) {
            const int k = k0 + (t >> 4);
            if (k < S_) {
                const float* src = edge_query + (((size_t)b * S_ + k) * S_ + q) * D_ + sub * 4;
                async_copy16(src, eqbuf + k * D_ + sub * 4);
            }
        }
    }
    // warm L2 for the phase-3 edge_value slice while we compute scores
    if (t < S_)
        __builtin_prefetch(edge_value + (((size_t)b * S_ + q) * S_ + t) * D_, 0, 1);

    async_wait_all();
    __syncthreads();

    // ---- Phase 1: scores; thread t handles key index k = t ----
    if (t < S_) {
        const int k = t;
        const float4* kv = (const float4*)(Kp + ((size_t)b * S_ + k) * HD_);
        const float4* ek = (const float4*)(ekbuf + k * D_);
        const float4* eq = (const float4*)(eqbuf + k * D_);
        const float4* qr = (const float4*)qrow;

        float t4 = 0.f;
        float tqk[H_] = {0.f, 0.f, 0.f, 0.f};
        float tqe[H_] = {0.f, 0.f, 0.f, 0.f};
        float tek[H_] = {0.f, 0.f, 0.f, 0.f};

        for (int d4 = 0; d4 < D_ / 4; ++d4) {         // 16 float4 chunks
            float4 e1 = ek[d4];
            float4 e2 = eq[d4];
            t4 += dot4(e2, e1);
            #pragma unroll
            for (int h = 0; h < H_; ++h) {
                float4 kvv = kv[h * (D_ / 4) + d4];
                float4 qv  = qr[h * (D_ / 4) + d4];
                tqk[h] += dot4(qv, kvv);
                tqe[h] += dot4(qv, e1);
                tek[h] += dot4(e2, kvv);
            }
        }

        const int  blk   = (q >= 3) ? (3 + 2 * ((q - 3) >> 1)) : 0;
        const bool keepm = (q < 3) || (k < 3) || ((k >= blk) && (k < blk + 2));
        const bool keep  = keepm && (graph[((size_t)b * S_ + q) * S_ + k] != 0);

        #pragma unroll
        for (int h = 0; h < H_; ++h) {
            float s = (tqk[h] + tqe[h] + tek[h] + t4) * SCALE_;
            sc[h][k] = keep ? s : NEG_;
        }
    }
    __syncthreads();

    // ---- Phase 2: softmax over k; wave w handles head w ----
    {
        const int h    = t >> 5;
        const int lane = t & 31;
        float m = -3.0e38f;
        for (int k = lane; k < S_; k += 32) m = fmaxf(m, sc[h][k]);
        #pragma unroll
        for (int off = 16; off > 0; off >>= 1) m = fmaxf(m, __shfl_xor(m, off));
        float ssum = 0.f;
        for (int k = lane; k < S_; k += 32) {
            float e = __expf(sc[h][k] - m);
            pr[h][k] = e;
            ssum += e;
        }
        #pragma unroll
        for (int off = 16; off > 0; off >>= 1) ssum += __shfl_xor(ssum, off);
        const float inv = 1.0f / ssum;
        for (int k = lane; k < S_; k += 32) pr[h][k] *= inv;
    }
    __syncthreads();

    // ---- Phase 3: out[h,d] = sum_k p[h,k] * (V[b,k,h,d] + edge_value[b,q,k,d]) ----
    {
        const int o0 = t;
        const int o1 = t + 128;
        const int h0 = o0 >> 6;          // 0 or 1
        const int h1 = o1 >> 6;          // 2 or 3
        const int d  = t & 63;           // same d for both outputs
        float a0 = 0.f, a1 = 0.f;
        const float* vbase  = Vp + (size_t)b * S_ * HD_;
        const float* evbase = edge_value + (((size_t)b * S_ + q) * S_) * D_;
        for (int k = 0; k < S_; ++k) {
            const float  ev = evbase[(size_t)k * D_ + d];
            const float* vr = vbase + (size_t)k * HD_;
            a0 += pr[h0][k] * (vr[o0] + ev);
            a1 += pr[h1][k] * (vr[o1] + ev);
        }
        Xa[(size_t)bq * HD_ + o0] = a0;
        Xa[(size_t)bq * HD_ + o1] = a1;
    }
}

// ---------------- host launcher ----------------
extern "C" void kernel_launch(void* const* d_in, const int* in_sizes, int n_in,
                              void* d_out, int out_size, void* d_ws, size_t ws_size,
                              hipStream_t stream) {
    const float* query      = (const float*)d_in[0];
    const float* key        = (const float*)d_in[1];
    const float* value      = (const float*)d_in[2];
    const int*   graph      = (const int*)  d_in[3];
    const float* edge_key   = (const float*)d_in[4];
    const float* edge_value = (const float*)d_in[5];
    const float* edge_query = (const float*)d_in[6];
    const float* Wq = (const float*)d_in[7];  const float* bq = (const float*)d_in[8];
    const float* Wk = (const float*)d_in[9];  const float* bk = (const float*)d_in[10];
    const float* Wv = (const float*)d_in[11]; const float* bv = (const float*)d_in[12];
    const float* Wo = (const float*)d_in[13]; const float* bo = (const float*)d_in[14];

    char*  ws        = (char*)d_ws;
    size_t projBytes = (size_t)R_ * HD_ * sizeof(float);   // 4,161,536 B (16B aligned)
    float* Qp = (float*)(ws + 0 * projBytes);
    float* Kp = (float*)(ws + 1 * projBytes);
    float* Vp = (float*)(ws + 2 * projBytes);
    float* Xa = (float*)(ws + 3 * projBytes);
    _Float16* Wqh = (_Float16*)(ws + 4 * projBytes);
    _Float16* Wkh = Wqh + HD_ * HD_;
    _Float16* Wvh = Wkh + HD_ * HD_;
    _Float16* Woh = Wvh + HD_ * HD_;

    const int nW = HD_ * HD_;
    cvt_f32_f16<<<(nW + 255) / 256, 256, 0, stream>>>(Wq, Wqh, nW);
    cvt_f32_f16<<<(nW + 255) / 256, 256, 0, stream>>>(Wk, Wkh, nW);
    cvt_f32_f16<<<(nW + 255) / 256, 256, 0, stream>>>(Wv, Wvh, nW);
    cvt_f32_f16<<<(nW + 255) / 256, 256, 0, stream>>>(Wo, Woh, nW);

    dim3 g(R_ / 16, HD_ / 64);   // 254 x 4 tiles (16 x 64 per wave)
    gemm_wmma<<<g, 32, 0, stream>>>(query, Wqh, bq, Qp);
    gemm_wmma<<<g, 32, 0, stream>>>(key,   Wkh, bk, Kp);
    gemm_wmma<<<g, 32, 0, stream>>>(value, Wvh, bv, Vp);

    attn_kernel<<<R_, 128, 0, stream>>>(Qp, Kp, Vp, graph,
                                        edge_key, edge_value, edge_query, Xa);

    gemm_wmma<<<g, 32, 0, stream>>>(Xa, Woh, bo, (float*)d_out);
}
